// NeighborAttention_19138374271379
// MI455X (gfx1250) — matrix operand order repaired and checked by reading, not compiled, not run on hardware
//
#include <hip/hip_runtime.h>

#define HD    256   // NUM_HIDDEN
#define K3H   768   // 3*H (bias-MLP input)
#define KIN   512   // NUM_IN
#define WAVES 4
#define BDIM  128   // 4 waves of 32

typedef __attribute__((ext_vector_type(16))) __bf16 v16bf;
typedef __attribute__((ext_vector_type(8)))  __bf16 v8bf;
typedef __attribute__((ext_vector_type(8)))  float  v8f;
typedef __attribute__((ext_vector_type(4)))  unsigned int u32x4;
typedef __attribute__((ext_vector_type(8)))  int i32x8;
typedef __attribute__((ext_vector_type(4)))  int i32x4;

#if defined(__HIP_DEVICE_COMPILE__) && __has_builtin(__builtin_amdgcn_tensor_load_to_lds) && __has_builtin(__builtin_amdgcn_s_wait_tensorcnt)
#define HAVE_TDM 1
#else
#define HAVE_TDM 0
#endif

__device__ __forceinline__ v8f wmma_bf16(v16bf a, v16bf b, v8f c) {
  // D = A(16x32 bf16) * B(32x16 bf16) + C(16x16 f32)
  return __builtin_amdgcn_wmma_f32_16x16x32_bf16(false, a, false, b, (short)0, c, false, false);
}

// A-fragment per ISA layout: lane half h owns K runs [k0+8h, k0+8h+7] and [k0+16+8h, ...+7]
__device__ __forceinline__ v16bf afrag_bf16(const __bf16* row, int k0, int half) {
  v8bf lo = *(const v8bf*)(row + k0 + 8*half);
  v8bf hi = *(const v8bf*)(row + k0 + 16 + 8*half);
  v16bf a;
#pragma unroll
  for (int i = 0; i < 8; ++i) { a[i] = lo[i]; a[i + 8] = hi[i]; }
  return a;
}

__device__ __forceinline__ v16bf bfrag_kn(const __bf16* col, int ld, int half) {
  v16bf b;
#pragma unroll
  for (int i = 0; i < 8; ++i) { b[i] = col[(8*half + i)*ld]; b[i + 8] = col[(16 + 8*half + i)*ld]; }
  return b;
}

__device__ __forceinline__ v16bf afrag_f32(const float* row, int k0, int half) {
  const float4* p1 = (const float4*)(row + k0 + 8*half);
  const float4* p2 = (const float4*)(row + k0 + 16 + 8*half);
  float4 x0 = p1[0], x1 = p1[1], y0 = p2[0], y1 = p2[1];
  v16bf a;
  a[0]=(__bf16)x0.x;  a[1]=(__bf16)x0.y;  a[2]=(__bf16)x0.z;  a[3]=(__bf16)x0.w;
  a[4]=(__bf16)x1.x;  a[5]=(__bf16)x1.y;  a[6]=(__bf16)x1.z;  a[7]=(__bf16)x1.w;
  a[8]=(__bf16)y0.x;  a[9]=(__bf16)y0.y;  a[10]=(__bf16)y0.z; a[11]=(__bf16)y0.w;
  a[12]=(__bf16)y1.x; a[13]=(__bf16)y1.y; a[14]=(__bf16)y1.z; a[15]=(__bf16)y1.w;
  return a;
}

#if HAVE_TDM
// Issue a TDM copy of one contiguous 16KB weight k-slice (32xHD bf16) into LDS.
// D# per CDNA5 ISA: 1-D tile, data_size=8B, tile_dim0=2048.
__device__ __forceinline__ void tdm_issue(__bf16* dstLds, const __bf16* src) {
  unsigned lds = (unsigned)(size_t)(void*)dstLds;                 // LDS byte offset (low 32 bits)
  unsigned long long ga = (unsigned long long)(size_t)(const void*)src;
  u32x4 g0;
  g0[0] = 1u;                                                     // count=1, user mode
  g0[1] = lds;                                                    // lds_addr
  g0[2] = (unsigned)ga;                                           // global_addr[31:0]
  g0[3] = (unsigned)((ga >> 32) & 0x01FFFFFFu) | (2u << 30);      // global_addr[56:32] | type=2
  i32x8 g1;
  g1[0] = (int)(3u << 16);        // wg_mask=0, data_size=3 (8B)
  g1[1] = (int)(0x800u << 16);    // atomic_barrier_addr=0, tensor_dim0.lo16=2048
  g1[2] = (int)(1u << 16);        // tensor_dim0.hi=0, tensor_dim1.lo16=1
  g1[3] = (int)(0x800u << 16);    // tensor_dim1.hi=0, tile_dim0=2048
  g1[4] = 0;                      // tile_dim1=0 (1-D), tile_dim2=0
  g1[5] = 2048;                   // tensor_dim0_stride[31:0]
  g1[6] = 0;                      // stride hi, tensor_dim1_stride.lo
  g1[7] = 0;
  i32x4 gz4; gz4[0] = 0; gz4[1] = 0; gz4[2] = 0; gz4[3] = 0;
  i32x8 gz8;
#pragma unroll
  for (int i = 0; i < 8; ++i) gz8[i] = 0;
  // clang-23 / amdgpu-toolchain 6-arg form: (g0, g1, g2, g3, g4, cpol)
  __builtin_amdgcn_tensor_load_to_lds(g0, g1, gz4, gz4, gz8, 0);
}
#endif

// Fallback: block-cooperative staging of a 32xHD bf16 slice (16 KB).
__device__ __forceinline__ void stageB(__bf16* Bs, const __bf16* src, int tid) {
  const v8bf* s = (const v8bf*)src;
  v8bf* d = (v8bf*)Bs;
#pragma unroll
  for (int i = 0; i < 8; ++i) d[tid + i*BDIM] = s[tid + i*BDIM];
}

// K-loop over NK 32-wide slices of weight W (K-major, HD cols), accumulating 16 N-tiles.
// TDM path: double-buffered LDS, copy of slice k+1 overlaps compute of slice k.
template <typename AF>
__device__ __forceinline__ void gemm_kloop(v8f (&acc)[16], __bf16* Bs0, __bf16* Bs1,
                                           const __bf16* W, int NK, int tid, int lane, AF afr)
{
#if HAVE_TDM
  if (tid < 32) tdm_issue(Bs0, W);
  for (int kt = 0; kt < NK; ++kt) {
    __bf16* Bcur = (kt & 1) ? Bs1 : Bs0;
    __bf16* Bnxt = (kt & 1) ? Bs0 : Bs1;
    if (tid < 32) {
      if (kt + 1 < NK) {
        tdm_issue(Bnxt, W + (size_t)(kt + 1) * 32 * HD);
        __builtin_amdgcn_s_wait_tensorcnt(1);   // slice kt complete; kt+1 still in flight
      } else {
        __builtin_amdgcn_s_wait_tensorcnt(0);
      }
    }
    __syncthreads();
    v16bf a = afr(kt);
    const __bf16* Brow = Bcur + (lane & 15);
    v16bf b0 = bfrag_kn(Brow, HD, lane >> 4);
#pragma unroll
    for (int t = 0; t < 16; ++t) {
      if (t < 15) {
        v16bf b1 = bfrag_kn(Brow + (t + 1)*16, HD, lane >> 4);   // prefetch next tile's B
        acc[t] = wmma_bf16(a, b0, acc[t]);
        b0 = b1;
      } else {
        acc[t] = wmma_bf16(a, b0, acc[t]);
      }
    }
    __syncthreads();
  }
#else
  for (int kt = 0; kt < NK; ++kt) {
    __syncthreads();
    stageB(Bs0, W + (size_t)kt * 32 * HD, tid);
    __syncthreads();
    v16bf a = afr(kt);
    const __bf16* Brow = Bs0 + (lane & 15);
    v16bf b0 = bfrag_kn(Brow, HD, lane >> 4);
#pragma unroll
    for (int t = 0; t < 16; ++t) {
      if (t < 15) {
        v16bf b1 = bfrag_kn(Brow + (t + 1)*16, HD, lane >> 4);
        acc[t] = wmma_bf16(a, b0, acc[t]);
        b0 = b1;
      } else {
        acc[t] = wmma_bf16(a, b0, acc[t]);
      }
    }
  }
#endif
}

// ---------------- Pass 1: per-edge bias MLP -> logits[E,4] ----------------
__global__ __launch_bounds__(BDIM) void k_edge_mlp(
    const unsigned short* __restrict__ hVb_u, const float* __restrict__ hE,
    const int* __restrict__ center,
    const unsigned short* __restrict__ B1t_u, const float* __restrict__ b1,
    const unsigned short* __restrict__ B2t_u, const float* __restrict__ b2,
    const unsigned short* __restrict__ B3t_u, const float* __restrict__ b3,
    float* __restrict__ logits)
{
  const __bf16* hVb = (const __bf16*)hVb_u;
  const __bf16* B1t = (const __bf16*)B1t_u;
  const __bf16* B2t = (const __bf16*)B2t_u;
  const __bf16* B3t = (const __bf16*)B3t_u;

  __shared__ __align__(32) __bf16 Bs0[32 * HD];        // 16 KB weight slice (buf 0)
  __shared__ __align__(32) __bf16 Bs1[32 * HD];        // 16 KB weight slice (buf 1)
  __shared__ __align__(32) __bf16 Ws[WAVES * 16 * HD]; // 32 KB per-wave activations

  const int tid  = threadIdx.x;
  const int wave = tid >> 5, lane = tid & 31;
  const int ln   = lane & 15, half = lane >> 4;
  const long e0  = ((long)blockIdx.x * WAVES + wave) * 16;

  const __bf16* aRowV = hVb + (long)center[e0 + ln] * HD;  // gathered node features
  const float*  aRowE = hE + (e0 + ln) * (long)KIN;        // edge features (fp32)
  __bf16*       myW   = Ws + wave * 16 * HD;
  const __bf16* myRow = myW + ln * HD;

  v8f acc[16];
  // ---- layer 1: [16,768] @ [768,256] ----
#pragma unroll
  for (int t = 0; t < 16; ++t) {
    float bv = b1[t*16 + ln];
#pragma unroll
    for (int v = 0; v < 8; ++v) acc[t][v] = bv;
  }
  gemm_kloop(acc, Bs0, Bs1, B1t, K3H/32, tid, lane, [&](int kt) {
    return (kt*32 < HD) ? afrag_bf16(aRowV, kt*32, half)
                        : afrag_f32(aRowE, kt*32 - HD, half);
  });
  // relu + restage as bf16 activations
#pragma unroll
  for (int t = 0; t < 16; ++t)
#pragma unroll
    for (int v = 0; v < 8; ++v)
      myW[(8*half + v)*HD + t*16 + ln] = (__bf16)fmaxf(acc[t][v], 0.0f);

  // ---- layer 2: [16,256] @ [256,256] ----
#pragma unroll
  for (int t = 0; t < 16; ++t) {
    float bv = b2[t*16 + ln];
#pragma unroll
    for (int v = 0; v < 8; ++v) acc[t][v] = bv;
  }
  gemm_kloop(acc, Bs0, Bs1, B2t, HD/32, tid, lane, [&](int kt) {
    return afrag_bf16(myRow, kt*32, half);
  });
  __syncthreads();
#pragma unroll
  for (int t = 0; t < 16; ++t)
#pragma unroll
    for (int v = 0; v < 8; ++v)
      myW[(8*half + v)*HD + t*16 + ln] = (__bf16)fmaxf(acc[t][v], 0.0f);
  __syncthreads();

  // ---- layer 3: [16,256] @ [256,16] (4 valid heads; 1/sqrt(64) folded into B3t) ----
  v8f acc3;
#pragma unroll
  for (int v = 0; v < 8; ++v) acc3[v] = 0.0f;
  for (int kt = 0; kt < HD/32; ++kt) {
    v16bf a = afrag_bf16(myRow, kt*32, half);
    v16bf b = bfrag_kn(B3t + kt*32*16 + ln, 16, half);   // tiny, straight from L2
    acc3 = wmma_bf16(a, b, acc3);
  }
  if (ln < 4) {
    float bb = 0.125f * b3[ln];
#pragma unroll
    for (int v = 0; v < 8; ++v)
      logits[(e0 + 8*half + v)*4 + ln] = acc3[v] + bb;
  }
}

// ---------------- Pass 2: scatter softmax pieces ----------------
__global__ void k_zero(float* __restrict__ p, long n) {
  long i = (long)blockIdx.x * 256 + threadIdx.x;
  if (i < n) p[i] = 0.0f;
}

__global__ void k_segmax(const float* __restrict__ logits, const int* __restrict__ center,
                         unsigned* __restrict__ segmax, long n4) {
  long i = (long)blockIdx.x * 256 + threadIdx.x;
  if (i >= n4) return;
  unsigned u = __float_as_uint(logits[i]);
  u = (u & 0x80000000u) ? ~u : (u | 0x80000000u);       // monotonic float->uint map
  atomicMax(&segmax[(long)center[i >> 2]*4 + (i & 3)], u);
}

__global__ void k_softmax_num(const float* __restrict__ logits, const int* __restrict__ center,
                              const unsigned* __restrict__ segmax, float* __restrict__ ex,
                              float* __restrict__ segsum, long n4) {
  long i = (long)blockIdx.x * 256 + threadIdx.x;
  if (i >= n4) return;
  long s = (long)center[i >> 2]*4 + (i & 3);
  unsigned u = segmax[s];
  u = (u & 0x80000000u) ? (u & 0x7fffffffu) : ~u;       // inverse map
  float e = expf(logits[i] - __uint_as_float(u));
  ex[i] = e;
  atomicAdd(&segsum[s], e);
}

// ---------------- Pass 3: V = h_E @ Wv.T, weighted scatter into agg ----------------
__global__ __launch_bounds__(BDIM) void k_value_scatter(
    const float* __restrict__ hE, const int* __restrict__ center,
    const unsigned short* __restrict__ Wvt_u, const float* __restrict__ ex,
    const float* __restrict__ segsum, float* __restrict__ agg)
{
  const __bf16* Wvt = (const __bf16*)Wvt_u;
  __shared__ __align__(32) __bf16 Bs0[32 * HD];
  __shared__ __align__(32) __bf16 Bs1[32 * HD];

  const int tid  = threadIdx.x;
  const int wave = tid >> 5, lane = tid & 31;
  const int ln   = lane & 15, half = lane >> 4;
  const long e0  = ((long)blockIdx.x * WAVES + wave) * 16;

  const float* aRowE = hE + (e0 + ln) * (long)KIN;
  int cm[8];
#pragma unroll
  for (int v = 0; v < 8; ++v) cm[v] = center[e0 + 8*half + v];

  v8f acc[16];
#pragma unroll
  for (int t = 0; t < 16; ++t)
#pragma unroll
    for (int v = 0; v < 8; ++v) acc[t][v] = 0.0f;

  gemm_kloop(acc, Bs0, Bs1, Wvt, KIN/32, tid, lane, [&](int kt) {
    return afrag_f32(aRowE, kt*32, half);
  });

  float att[8];
#pragma unroll
  for (int t = 0; t < 16; ++t) {
    int head = t >> 2;                                   // 16-col tile lies within one head
    if ((t & 3) == 0) {
#pragma unroll
      for (int v = 0; v < 8; ++v) {
        long e = e0 + 8*half + v;
        att[v] = ex[e*4 + head] / segsum[(long)cm[v]*4 + head];
      }
    }
#pragma unroll
    for (int v = 0; v < 8; ++v)
      atomicAdd(&agg[(long)cm[v]*HD + t*16 + ln], att[v] * acc[t][v]);
  }
}

// ---------------- Pass 4: h_out = agg @ Wo.T ----------------
__global__ __launch_bounds__(BDIM) void k_out_gemm(
    const float* __restrict__ agg, const unsigned short* __restrict__ Wot_u,
    float* __restrict__ out, int Mtiles)
{
  const __bf16* Wot = (const __bf16*)Wot_u;
  __shared__ __align__(32) __bf16 Bs0[32 * HD];
  __shared__ __align__(32) __bf16 Bs1[32 * HD];

  const int tid  = threadIdx.x;
  const int wave = tid >> 5, lane = tid & 31;
  const int ln   = lane & 15, half = lane >> 4;
  int r  = blockIdx.x * WAVES + wave;
  int rc = (r < Mtiles) ? r : (Mtiles - 1);              // keep barriers uniform
  const float* aRow = agg + ((long)rc*16 + ln) * HD;

  v8f acc[16];
#pragma unroll
  for (int t = 0; t < 16; ++t)
#pragma unroll
    for (int v = 0; v < 8; ++v) acc[t][v] = 0.0f;

  gemm_kloop(acc, Bs0, Bs1, Wot, HD/32, tid, lane, [&](int kt) {
    return afrag_f32(aRow, kt*32, half);
  });

  if (r < Mtiles) {
#pragma unroll
    for (int t = 0; t < 16; ++t)
#pragma unroll
      for (int v = 0; v < 8; ++v)
        out[((long)rc*16 + 8*half + v)*HD + t*16 + ln] = acc[t][v];
  }
}

// ---------------- Pass 0: weight conversion / transposition ----------------
__global__ void k_convert(const float* __restrict__ src, unsigned short* __restrict__ dst, long n) {
  long i = (long)blockIdx.x * 256 + threadIdx.x;
  if (i < n) ((__bf16*)dst)[i] = (__bf16)src[i];
}

// dst[c*rows + r] = src[r*cols + c]  (K-major weight for contiguous B-fragments)
__global__ void k_convertT(const float* __restrict__ src, unsigned short* __restrict__ dst,
                           int rows, int cols) {
  long i = (long)blockIdx.x * 256 + threadIdx.x;
  if (i >= (long)rows * cols) return;
  int r = (int)(i / cols), c = (int)(i % cols);
  ((__bf16*)dst)[(long)c*rows + r] = (__bf16)src[i];
}

// B3t[k][n] = (n<4) ? B3_w[n][k]/8 : 0   (pad heads to N=16, fold 1/sqrt(64))
__global__ void k_b3(const float* __restrict__ B3w, unsigned short* __restrict__ dst) {
  int i = blockIdx.x * 256 + threadIdx.x;
  if (i >= HD * 16) return;
  int k = i >> 4, n = i & 15;
  float v = (n < 4) ? B3w[n*HD + k] * 0.125f : 0.0f;
  ((__bf16*)dst)[i] = (__bf16)v;
}

extern "C" void kernel_launch(void* const* d_in, const int* in_sizes, int n_in,
                              void* d_out, int out_size, void* d_ws, size_t ws_size,
                              hipStream_t stream) {
  const float* h_V = (const float*)d_in[0];
  const float* h_E = (const float*)d_in[1];
  const int*   center = (const int*)d_in[2];
  // d_in[3] = batch_id (unused)
  const float* B1w = (const float*)d_in[4];
  const float* b1  = (const float*)d_in[5];
  const float* B2w = (const float*)d_in[6];
  const float* b2  = (const float*)d_in[7];
  const float* B3w = (const float*)d_in[8];
  const float* b3  = (const float*)d_in[9];
  const float* Wv  = (const float*)d_in[10];
  const float* Wo  = (const float*)d_in[11];

  const long N = (long)in_sizes[0] / HD;   // 20000
  const long E = (long)in_sizes[2];        // 400000

  char* wsb = (char*)d_ws;
  size_t off = 0;
  auto carve = [&](size_t bytes) -> char* {
    char* p = wsb + off;
    off += (bytes + 255) & ~(size_t)255;
    return p;
  };
  unsigned short* hVb = (unsigned short*)carve((size_t)N * HD * 2);
  unsigned short* B1t = (unsigned short*)carve((size_t)K3H * HD * 2);
  unsigned short* B2t = (unsigned short*)carve((size_t)HD * HD * 2);
  unsigned short* B3t = (unsigned short*)carve((size_t)HD * 16 * 2);
  unsigned short* Wvt = (unsigned short*)carve((size_t)KIN * HD * 2);
  unsigned short* Wot = (unsigned short*)carve((size_t)HD * HD * 2);
  float*    logits = (float*)carve((size_t)E * 4 * 4);
  float*    ex     = (float*)carve((size_t)E * 4 * 4);
  unsigned* segmax = (unsigned*)carve((size_t)N * 4 * 4);
  float*    segsum = (float*)carve((size_t)N * 4 * 4);
  float*    agg    = (float*)carve((size_t)N * HD * 4);

  // Pass 0: conversions
  k_convert<<<(int)((N*HD + 255)/256), 256, 0, stream>>>(h_V, hVb, N*HD);
  k_convertT<<<(HD*K3H + 255)/256, 256, 0, stream>>>(B1w, B1t, HD, K3H);
  k_convertT<<<(HD*HD + 255)/256, 256, 0, stream>>>(B2w, B2t, HD, HD);
  k_convertT<<<(HD*KIN + 255)/256, 256, 0, stream>>>(Wv, Wvt, HD, KIN);
  k_convertT<<<(HD*HD + 255)/256, 256, 0, stream>>>(Wo, Wot, HD, HD);
  k_b3<<<(HD*16 + 255)/256, 256, 0, stream>>>(B3w, B3t);

  // zero segmax+segsum+agg (carved contiguously; zero the whole span)
  long zn = (long)(((char*)agg + (size_t)N*HD*4) - (char*)segmax) / 4;
  k_zero<<<(int)((zn + 255)/256), 256, 0, stream>>>((float*)segmax, zn);

  // Pass 1: edge MLP -> logits
  k_edge_mlp<<<(int)(E / (WAVES*16)), BDIM, 0, stream>>>(hVb, h_E, center,
                                                         B1t, b1, B2t, b2, B3t, b3, logits);
  // Pass 2: scatter softmax statistics
  k_segmax<<<(int)((E*4 + 255)/256), 256, 0, stream>>>(logits, center, segmax, E*4);
  k_softmax_num<<<(int)((E*4 + 255)/256), 256, 0, stream>>>(logits, center, segmax, ex, segsum, E*4);

  // Pass 3: values + weighted scatter-add
  k_value_scatter<<<(int)(E / (WAVES*16)), BDIM, 0, stream>>>(h_E, center, Wvt, ex, segsum, agg);

  // Pass 4: output projection
  int Mtiles = (int)(N / 16);
  k_out_gemm<<<(Mtiles + WAVES - 1)/WAVES, BDIM, 0, stream>>>(agg, Wot, (float*)d_out, Mtiles);
}
